// TextEncoder_14078902796503
// MI455X (gfx1250) — compile-verified, matrix operand order checked by reading, not run on hardware
//
#include <hip/hip_runtime.h>
#include <hip/hip_bf16.h>

// ---------------------------------------------------------------------------
// MI455X (gfx1250) implementation.
// All GEMMs use v_wmma_f32_16x16x32_bf16 (wave32, 16x16 tiles, K=32 chunks).
// A operands converted f32->bf16 on device; B operands stored transposed
// (N x K, row-major bf16) so both A and B fragments are contiguous 16-byte
// LDS reads matching the CDNA5 16-bit operand VGPR layout:
//   lanes 0-15  : row M=lane,    K = {0..7} then {16..23}   (two b128 loads)
//   lanes 16-31 : row M=lane-16, K = {8..15} then {24..31}
// Tiles staged global->LDS via GLOBAL_LOAD_ASYNC_TO_LDS_B128 when the
// toolchain exposes the builtin (ASYNCcnt path), else via uint4 copies.
// ---------------------------------------------------------------------------

#if __has_builtin(__builtin_amdgcn_global_load_async_to_lds_b128) && \
    __has_builtin(__builtin_amdgcn_s_wait_asynccnt)
#define HAVE_ASYNC_LDS 1
#else
#define HAVE_ASYNC_LDS 0
#endif

typedef __attribute__((ext_vector_type(4)))  int    v4i;
typedef __attribute__((ext_vector_type(8)))  __bf16 bf16x8;
typedef __attribute__((ext_vector_type(16))) __bf16 v16bf;
typedef __attribute__((ext_vector_type(8)))  float  v8f;

// pointer-to-AS1(v4i) / pointer-to-AS3(v4i), matching the builtin's signature
typedef __attribute__((address_space(1))) v4i* gbl_v4i_ptr;
typedef __attribute__((address_space(3))) v4i* lds_v4i_ptr;

__device__ __forceinline__ unsigned short f32_to_bf16u(float f) {
    unsigned int u = __float_as_uint(f);
    u += 0x7FFFu + ((u >> 16) & 1u);        // round-to-nearest-even
    return (unsigned short)(u >> 16);
}
__device__ __forceinline__ float sigmoidf(float x) { return 1.0f / (1.0f + expf(-x)); }
__device__ __forceinline__ float eluf(float x)     { return x > 0.f ? x : (expf(x) - 1.0f); }

__device__ __forceinline__ v16bf load_frag(const unsigned short* base) {
    bf16x8 lo = *(const bf16x8*)base;
    bf16x8 hi = *(const bf16x8*)(base + 16);
    return __builtin_shufflevector(lo, hi, 0,1,2,3,4,5,6,7,8,9,10,11,12,13,14,15);
}

// ---------------------------------------------------------------------------
// Batched GEMM: C[z] = A[z] (MxK, bf16, row-major, stride lda)
//                    @ Bt[z]^T (Bt is NxK bf16 row-major, stride ldb)
// C is f32 (ldc). acc=1 -> C += result. Requires M%128==0, N%64==0, K%32==0.
// Block: 256 threads = 8 waves (4 M x 2 N); block tile 128x64;
// wave tile 32x32 -> 4 WMMAs per K-chunk per wave (2 A frags + 2 B frags).
// ---------------------------------------------------------------------------
__global__ __launch_bounds__(256)
void gemm_bf16_wmma(const unsigned short* __restrict__ A, long long lda, long long sA,
                    const unsigned short* __restrict__ Bt, long long ldb, long long sB,
                    float* __restrict__ C, long long ldc, long long sC,
                    int M, int N, int K, int acc)
{
    __shared__ __align__(16) unsigned short As[128 * 32];
    __shared__ __align__(16) unsigned short Bs[64 * 32];

    const int tid   = threadIdx.x;
    const int lane  = tid & 31;
    const int wv    = tid >> 5;
    const int waveM = wv >> 1;            // 0..3 -> rows waveM*32
    const int waveN = wv & 1;             // 0..1 -> cols waveN*32
    const long long m0 = (long long)blockIdx.y * 128;
    const long long n0 = (long long)blockIdx.x * 64;
    const long long z  = blockIdx.z;
    A  += z * sA;  Bt += z * sB;  C += z * sC;

    v8f c00 = {}, c01 = {}, c10 = {}, c11 = {};

    const int ldRow = tid >> 2;           // 0..63
    const int ldCol = (tid & 3) * 8;      // 0,8,16,24

    for (int k0 = 0; k0 < K; k0 += 32) {
        const unsigned short* gA0 = &A[(m0 + ldRow) * lda + k0 + ldCol];
        const unsigned short* gA1 = &A[(m0 + 64 + ldRow) * lda + k0 + ldCol];
        const unsigned short* gB  = &Bt[(n0 + ldRow) * ldb + k0 + ldCol];
#if HAVE_ASYNC_LDS
        __builtin_amdgcn_global_load_async_to_lds_b128(
            (gbl_v4i_ptr)gA0, (lds_v4i_ptr)&As[ldRow * 32 + ldCol], 0, 0);
        __builtin_amdgcn_global_load_async_to_lds_b128(
            (gbl_v4i_ptr)gA1, (lds_v4i_ptr)&As[(64 + ldRow) * 32 + ldCol], 0, 0);
        __builtin_amdgcn_global_load_async_to_lds_b128(
            (gbl_v4i_ptr)gB,  (lds_v4i_ptr)&Bs[ldRow * 32 + ldCol], 0, 0);
        __builtin_amdgcn_s_wait_asynccnt(0);
#else
        *(uint4*)&As[ldRow * 32 + ldCol]        = *(const uint4*)gA0;
        *(uint4*)&As[(64 + ldRow) * 32 + ldCol] = *(const uint4*)gA1;
        *(uint4*)&Bs[ldRow * 32 + ldCol]        = *(const uint4*)gB;
#endif
        if (k0 + 32 < K) {   // global_prefetch_b8: pull next tiles toward L2
            __builtin_prefetch(gA0 + 32, 0, 1);
            __builtin_prefetch(gA1 + 32, 0, 1);
            __builtin_prefetch(gB + 32, 0, 1);
        }
        __syncthreads();

        const int r  = lane & 15;
        const int kb = (lane < 16) ? 0 : 8;

        v16bf a0 = load_frag(&As[(waveM * 32 + r) * 32 + kb]);
        v16bf a1 = load_frag(&As[(waveM * 32 + 16 + r) * 32 + kb]);
        v16bf b0 = load_frag(&Bs[(waveN * 32 + r) * 32 + kb]);
        v16bf b1 = load_frag(&Bs[(waveN * 32 + 16 + r) * 32 + kb]);

        c00 = __builtin_amdgcn_wmma_f32_16x16x32_bf16(false, a0, false, b0, (short)0, c00, false, false);
        c01 = __builtin_amdgcn_wmma_f32_16x16x32_bf16(false, a0, false, b1, (short)0, c01, false, false);
        c10 = __builtin_amdgcn_wmma_f32_16x16x32_bf16(false, a1, false, b0, (short)0, c10, false, false);
        c11 = __builtin_amdgcn_wmma_f32_16x16x32_bf16(false, a1, false, b1, (short)0, c11, false, false);
        __syncthreads();
    }

    // C/D layout: VGPR v -> M = v (lanes 0-15) or 8+v (lanes 16-31); N = lane&15
    const int r = lane & 15;
    const long long mB0 = m0 + waveM * 32 + ((lane < 16) ? 0 : 8);
    const long long mB1 = mB0 + 16;
    const long long nc0 = n0 + waveN * 32 + r;
#pragma unroll
    for (int v = 0; v < 8; ++v) {
        long long i00 = (mB0 + v) * ldc + nc0;
        long long i10 = (mB1 + v) * ldc + nc0;
        if (acc) {
            C[i00]      += c00[v];  C[i00 + 16] += c01[v];
            C[i10]      += c10[v];  C[i10 + 16] += c11[v];
        } else {
            C[i00]      = c00[v];   C[i00 + 16] = c01[v];
            C[i10]      = c10[v];   C[i10 + 16] = c11[v];
        }
    }
}

// ---------------------------------------------------------------------------
// Elementwise / helper kernels
// ---------------------------------------------------------------------------
__global__ void cvt_f32_bf16(const float* __restrict__ in, unsigned short* __restrict__ out, long long n) {
    long long i = (long long)blockIdx.x * blockDim.x + threadIdx.x;
    if (i < n) out[i] = f32_to_bf16u(in[i]);
}

// out[z][c][r] = bf16(in[z][r][c])   (batched transpose + convert)
__global__ void transpose_cvt(const float* __restrict__ in, unsigned short* __restrict__ out,
                              int R, int C, long long inS, long long outS) {
    long long z   = blockIdx.z;
    long long idx = (long long)blockIdx.x * blockDim.x + threadIdx.x;
    if (idx >= (long long)R * C) return;
    int r = (int)(idx / C), c = (int)(idx % C);
    out[z * outS + (long long)c * R + r] = f32_to_bf16u(in[z * inS + idx]);
}

// f1 = Wh @ a[:D], f2 = Wh @ a[D:]; one wave per row
__global__ __launch_bounds__(256)
void f12_kernel(const float* __restrict__ Wh, const float* __restrict__ avec,
                float* __restrict__ f1, float* __restrict__ f2, int D, int rows) {
    int wave = blockIdx.x * 8 + (threadIdx.x >> 5);
    if (wave >= rows) return;
    int lane = threadIdx.x & 31;
    const float* row = Wh + (long long)wave * D;
    float s1 = 0.f, s2 = 0.f;
    for (int k = lane; k < D; k += 32) { float w = row[k]; s1 += w * avec[k]; s2 += w * avec[D + k]; }
    for (int off = 16; off; off >>= 1) { s1 += __shfl_xor(s1, off, 32); s2 += __shfl_xor(s2, off, 32); }
    if (lane == 0) { f1[wave] = s1; f2[wave] = s2; }
}

// masked leaky-relu softmax row -> bf16 attention; one wave per (graph,row), T=128
__global__ __launch_bounds__(256)
void att_softmax(const float* __restrict__ f1, const float* __restrict__ f2,
                 const int* __restrict__ adj, unsigned short* __restrict__ att,
                 int T, int rowsTotal) {
    int wave = blockIdx.x * 8 + (threadIdx.x >> 5);
    if (wave >= rowsTotal) return;
    int lane = threadIdx.x & 31;
    int g = wave / T, i = wave % T;
    float fi = f1[wave];
    float e[4];
#pragma unroll
    for (int it = 0; it < 4; ++it) {
        int j = lane + it * 32;
        float s = fi + f2[(long long)g * T + j];
        s = (s > 0.f) ? s : 0.2f * s;                      // leaky_relu(0.2)
        int m = adj[((long long)g * T + i) * T + j];
        e[it] = (m > 0) ? s : -1.0e9f;
    }
    float mx = fmaxf(fmaxf(e[0], e[1]), fmaxf(e[2], e[3]));
    for (int off = 16; off; off >>= 1) mx = fmaxf(mx, __shfl_xor(mx, off, 32));
    float p[4], sum = 0.f;
#pragma unroll
    for (int it = 0; it < 4; ++it) { p[it] = expf(e[it] - mx); sum += p[it]; }
    for (int off = 16; off; off >>= 1) sum += __shfl_xor(sum, off, 32);
    float inv = 1.0f / sum;
#pragma unroll
    for (int it = 0; it < 4; ++it)
        att[((long long)g * T + i) * T + lane + it * 32] = f32_to_bf16u(p[it] * inv);
}

// hcat[row, hd*D + c] = bf16(elu(in[row, c]))
__global__ void elu_cat(const float* __restrict__ in, unsigned short* __restrict__ hcat,
                        int hd, long long n, int D, int HC) {
    long long i = (long long)blockIdx.x * blockDim.x + threadIdx.x;
    if (i >= n) return;
    long long row = i / D; int c = (int)(i % D);
    hcat[row * HC + (long long)hd * D + c] = f32_to_bf16u(eluf(in[i]));
}

__global__ void elu_f32(const float* __restrict__ in, float* __restrict__ out, long long n) {
    long long i = (long long)blockIdx.x * blockDim.x + threadIdx.x;
    if (i < n) out[i] = eluf(in[i]);
}

__global__ void add_inplace(float* __restrict__ a, const float* __restrict__ b, long long n) {
    long long i = (long long)blockIdx.x * blockDim.x + threadIdx.x;
    if (i < n) a[i] += b[i];
}

__global__ void bias_sum(const float* __restrict__ a, const float* __restrict__ b,
                         float* __restrict__ out, int n) {
    int i = blockIdx.x * blockDim.x + threadIdx.x;
    if (i < n) out[i] = a[i] + b[i];
}

__global__ void zero_f32(float* p, long long n) {
    long long i = (long long)blockIdx.x * blockDim.x + threadIdx.x;
    if (i < n) p[i] = 0.f;
}
__global__ void zero_u16(unsigned short* p, long long n) {
    long long i = (long long)blockIdx.x * blockDim.x + threadIdx.x;
    if (i < n) p[i] = 0;
}

// LSTM gate fuse: g (128 x 3072 f32, gates [i|f|g|o]), updates c, writes bf16 h
// for the next step's GEMM, and captures fw (t==sl-1) or bw (t==0 on reverse).
__global__ void lstm_gate(const float* __restrict__ g, const float* __restrict__ badd,
                          float* __restrict__ c, unsigned short* __restrict__ hbf,
                          const int* __restrict__ sl, float* __restrict__ cap,
                          int t, int reverse, int T, int D) {
    int idx = blockIdx.x * blockDim.x + threadIdx.x;
    if (idx >= 128 * D) return;
    int b = idx / D, j = idx % D;
    long long base = (long long)b * 4 * D;
    float gi = g[base + j]          + badd[j];
    float gf = g[base + D + j]      + badd[D + j];
    float gg = g[base + 2 * D + j]  + badd[2 * D + j];
    float go = g[base + 3 * D + j]  + badd[3 * D + j];
    float cn = sigmoidf(gf) * c[idx] + sigmoidf(gi) * tanhf(gg);
    float h  = sigmoidf(go) * tanhf(cn);
    c[idx] = cn;
    hbf[idx] = f32_to_bf16u(h);
    if (!reverse) {
        int s = sl[b]; s = s < 1 ? 1 : (s > T ? T : s);
        if (t == s - 1) cap[idx] = h;
    } else {
        if (t == 0) cap[idx] = h;
    }
}

// Final classifier heads: one wave per (batch, head, class)
__global__ __launch_bounds__(256)
void heads_kernel(const float* __restrict__ fw, const float* __restrict__ bw,
                  const float* __restrict__ senW, const float* __restrict__ senb,
                  const float* __restrict__ emoW, const float* __restrict__ emob,
                  float* __restrict__ out, int D, int C) {
    int wave = blockIdx.x * 8 + (threadIdx.x >> 5);
    if (wave >= 128 * 2 * C) return;
    int lane = threadIdx.x & 31;
    int b = wave / (2 * C), q = wave % (2 * C);
    int hh = q / C, cc = q % C;
    const float* W = hh ? emoW : senW;
    float s = 0.f;
    for (int k = lane; k < 2 * D; k += 32) {
        float v = (k < D) ? fw[(long long)b * D + k] : bw[(long long)b * D + (k - D)];
        s += v * W[(long long)k * C + cc];
    }
    for (int off = 16; off; off >>= 1) s += __shfl_xor(s, off, 32);
    if (lane == 0) out[(long long)hh * 128 * C + (long long)b * C + cc] = s + (hh ? emob[cc] : senb[cc]);
}

// ---------------------------------------------------------------------------
// Host orchestration
// ---------------------------------------------------------------------------
static inline size_t alignUp256(size_t v) { return (v + 255) & ~size_t(255); }

extern "C" void kernel_launch(void* const* d_in, const int* in_sizes, int n_in,
                              void* d_out, int out_size, void* d_ws, size_t ws_size,
                              hipStream_t stream) {
    (void)in_sizes; (void)n_in; (void)out_size; (void)ws_size;

    const int DU = 128, T = 128, F = 768, D = 768, H = 4, HC = 3072, C = 7;
    const long long ROWS = (long long)DU * T;        // 16384
    const long long S    = ROWS * F;                 // 12,582,912

    const float* x       = (const float*)d_in[0];
    const int*   adj     = (const int*)d_in[1];
    const int*   sl      = (const int*)d_in[2];
    const float* gatW    = (const float*)d_in[3];
    const float* gatA    = (const float*)d_in[4];
    const float* gatOW   = (const float*)d_in[5];
    const float* gatOA   = (const float*)d_in[6];
    const float* wih_f   = (const float*)d_in[7];
    const float* whh_f   = (const float*)d_in[8];
    const float* bih_f   = (const float*)d_in[9];
    const float* bhh_f   = (const float*)d_in[10];
    const float* wih_b   = (const float*)d_in[11];
    const float* whh_b   = (const float*)d_in[12];
    const float* bih_b   = (const float*)d_in[13];
    const float* bhh_b   = (const float*)d_in[14];
    const float* senW    = (const float*)d_in[15];
    const float* senb    = (const float*)d_in[16];
    const float* emoW    = (const float*)d_in[17];
    const float* emob    = (const float*)d_in[18];
    float* out = (float*)d_out;

    // workspace carve (~330 MB assumed available)
    char* p = (char*)d_ws;
    auto carve = [&](size_t bytes) -> void* { void* r = (void*)p; p += alignUp256(bytes); return r; };
    float*          h_f32  = (float*)carve(S * 4);
    float*          tmpC   = (float*)carve(S * 4);
    float*          tmpC2  = (float*)carve(S * 4);
    unsigned short* h_bf   = (unsigned short*)carve(S * 2);
    unsigned short* tmpT   = (unsigned short*)carve(S * 2);
    unsigned short* hcat   = (unsigned short*)carve(ROWS * HC * 2);
    unsigned short* attb   = (unsigned short*)carve((long long)DU * T * T * 2);
    float*          f1     = (float*)carve(ROWS * 4);
    float*          f2     = (float*)carve(ROWS * 4);
    unsigned short* wT     = (unsigned short*)carve((long long)D * HC * 2);
    unsigned short* wihT   = (unsigned short*)carve((long long)HC * F * 2);
    unsigned short* whhT   = (unsigned short*)carve((long long)HC * D * 2);
    float*          badd   = (float*)carve(HC * 4);
    float*          gbuf   = (float*)carve((long long)128 * HC * 4);
    unsigned short* hstep  = (unsigned short*)carve((long long)128 * D * 2);
    float*          cbuf   = (float*)carve((long long)128 * D * 4);
    float*          fwb    = (float*)carve((long long)128 * D * 4);
    float*          bwb    = (float*)carve((long long)128 * D * 4);

    const int blkS  = (int)((S + 255) / 256);
    const int blkW8 = (int)((ROWS + 7) / 8);        // 1 wave per row, 8 waves/block

    // ---------------- GAT layers ----------------
    const float* hsrc = x;
    for (int l = 0; l < 2; ++l) {
        cvt_f32_bf16<<<blkS, 256, 0, stream>>>(hsrc, h_bf, S);
        for (int hd = 0; hd < H; ++hd) {
            const float* Wl = gatW + (long long)(l * H + hd) * F * D;
            transpose_cvt<<<dim3((F * D + 255) / 256, 1, 1), 256, 0, stream>>>(Wl, wT, F, D, 0, 0);
            // Wh = h @ W : (16384 x 768) @ (768 x 768)
            gemm_bf16_wmma<<<dim3(D / 64, (unsigned)(ROWS / 128), 1), 256, 0, stream>>>(
                h_bf, F, 0, wT, F, 0, tmpC, D, 0, (int)ROWS, D, F, 0);
            f12_kernel<<<blkW8, 256, 0, stream>>>(tmpC, gatA + (long long)(l * H + hd) * 2 * D,
                                                  f1, f2, D, (int)ROWS);
            // per-graph Wh^T for the att@Wh GEMM
            transpose_cvt<<<dim3((T * D + 255) / 256, 1, DU), 256, 0, stream>>>(
                tmpC, tmpT, T, D, (long long)T * D, (long long)D * T);
            att_softmax<<<blkW8, 256, 0, stream>>>(f1, f2, adj, attb, T, (int)ROWS);
            // headout = att @ Wh, batched over 128 graphs
            gemm_bf16_wmma<<<dim3(D / 64, T / 128, DU), 256, 0, stream>>>(
                attb, T, (long long)T * T, tmpT, T, (long long)D * T,
                tmpC2, D, (long long)T * D, T, D, T, 0);
            elu_cat<<<blkS, 256, 0, stream>>>(tmpC2, hcat, hd, S, D, HC);
        }
        // output head on concat(3072)
        transpose_cvt<<<dim3((HC * D + 255) / 256, 1, 1), 256, 0, stream>>>(
            gatOW + (long long)l * HC * D, wT, HC, D, 0, 0);
        gemm_bf16_wmma<<<dim3(D / 64, (unsigned)(ROWS / 128), 1), 256, 0, stream>>>(
            hcat, HC, 0, wT, HC, 0, tmpC, D, 0, (int)ROWS, D, HC, 0);
        f12_kernel<<<blkW8, 256, 0, stream>>>(tmpC, gatOA + (long long)l * 2 * D, f1, f2, D, (int)ROWS);
        transpose_cvt<<<dim3((T * D + 255) / 256, 1, DU), 256, 0, stream>>>(
            tmpC, tmpT, T, D, (long long)T * D, (long long)D * T);
        att_softmax<<<blkW8, 256, 0, stream>>>(f1, f2, adj, attb, T, (int)ROWS);
        gemm_bf16_wmma<<<dim3(D / 64, T / 128, DU), 256, 0, stream>>>(
            attb, T, (long long)T * T, tmpT, T, (long long)D * T,
            tmpC2, D, (long long)T * D, T, D, T, 0);
        elu_f32<<<blkS, 256, 0, stream>>>(tmpC2, h_f32, S);
        hsrc = h_f32;
    }

    // residual: h = x + h, then bf16 for LSTM input GEMMs
    add_inplace<<<blkS, 256, 0, stream>>>(h_f32, x, S);
    cvt_f32_bf16<<<blkS, 256, 0, stream>>>(h_f32, h_bf, S);

    // ---------------- BiLSTM ----------------
    for (int dir = 0; dir < 2; ++dir) {
        const float* wih = dir ? wih_b : wih_f;
        const float* whh = dir ? whh_b : whh_f;
        const float* bih = dir ? bih_b : bih_f;
        const float* bhh = dir ? bhh_b : bhh_f;
        float* cap = dir ? bwb : fwb;

        // B = Wih^T means Bt = Wih (4D x D row-major) -> convert only
        cvt_f32_bf16<<<(int)(((long long)HC * F + 255) / 256), 256, 0, stream>>>(wih, wihT, (long long)HC * F);
        cvt_f32_bf16<<<(int)(((long long)HC * D + 255) / 256), 256, 0, stream>>>(whh, whhT, (long long)HC * D);
        bias_sum<<<(HC + 255) / 256, 256, 0, stream>>>(bih, bhh, badd, HC);
        zero_f32<<<(128 * D + 255) / 256, 256, 0, stream>>>(cbuf, (long long)128 * D);
        zero_u16<<<(128 * D + 255) / 256, 256, 0, stream>>>(hstep, (long long)128 * D);

        for (int s = 0; s < T; ++s) {
            int t = dir ? (T - 1 - s) : s;
            // g = x_t @ Wih^T   (rows of h_bf strided by T*F)
            gemm_bf16_wmma<<<dim3(HC / 64, 1, 1), 256, 0, stream>>>(
                h_bf + (long long)t * F, (long long)T * F, 0,
                wihT, F, 0, gbuf, HC, 0, 128, HC, F, 0);
            // g += h @ Whh^T
            gemm_bf16_wmma<<<dim3(HC / 64, 1, 1), 256, 0, stream>>>(
                hstep, D, 0, whhT, D, 0, gbuf, HC, 0, 128, HC, D, 1);
            lstm_gate<<<(128 * D + 255) / 256, 256, 0, stream>>>(
                gbuf, badd, cbuf, hstep, sl, cap, t, dir, T, D);
        }
    }

    // ---------------- heads ----------------
    heads_kernel<<<(128 * 2 * C + 7) / 8, 256, 0, stream>>>(
        fwb, bwb, senW, senb, emoW, emob, out, D, C);
}
